// MultiHeadAttn_17076789969053
// MI455X (gfx1250) — matrix operand order repaired and checked by reading, not compiled, not run on hardware
//
#include <hip/hip_runtime.h>

// ---------------------------------------------------------------------------
// MI455X (gfx1250) multi-head attention, WMMA f16 path (f32 accumulate).
// weight cvt -> per-head projections (V stored transposed) ->
// double-buffered flash attention with async-to-LDS DMA ->
// double-buffered output projection with async-to-LDS DMA.
// ---------------------------------------------------------------------------

#define S_LEN 2048
#define NB    4
#define NH    16
#define HD    64
#define DM    1024
#define LDP   72       // padded LDS row stride (halves): 144B row, 16B pad
#define NKBLK (S_LEN / 64)

typedef __attribute__((ext_vector_type(16))) _Float16 v16h;
typedef __attribute__((ext_vector_type(8)))  _Float16 v8h;
typedef __attribute__((ext_vector_type(8)))  float    v8f;

__device__ __forceinline__ v8f wmma16(v16h a, v16h b, v8f c) {
  // emits v_wmma_f32_16x16x32_f16
  return __builtin_amdgcn_wmma_f32_16x16x32_f16(false, a, false, b, (short)0, c,
                                                false, false);
}

// A fragment, 16x32 f16 (MxK), row-major source with stride ld (halves).
// lane L(0-15) = row L, K in {0..7,16..23}; lane L+16: K {8..15,24..31}
__device__ __forceinline__ v16h load_a_frag(const _Float16* src, int ld, int lane) {
  const int row = lane & 15;
  const int kb  = (lane >> 4) << 3;
  v16h a;
#pragma unroll
  for (int h = 0; h < 16; ++h) {
    const int k = ((h >> 3) << 4) + kb + (h & 7);
    a[h] = src[row * ld + k];
  }
  return a;
}

// B fragment, 32x16 (KxN), from an N-major source: B(k,n) = src[n*ld + k].
// lane L(0-15): K in {0..7,16..23}; lane L+16: K {8..15,24..31}; N = lane&15.
__device__ __forceinline__ v16h load_bt_frag(const _Float16* src, int ld, int lane) {
  const int col = lane & 15;
  const int kb  = (lane >> 4) << 3;
  v16h b;
#pragma unroll
  for (int h = 0; h < 16; ++h) b[h] = src[col * ld + ((h >> 3) << 4) + kb + (h & 7)];
  return b;
}

__device__ __forceinline__ float half_shfl_max(float v) {
  v = fmaxf(v, __shfl_xor(v, 1, 32));
  v = fmaxf(v, __shfl_xor(v, 2, 32));
  v = fmaxf(v, __shfl_xor(v, 4, 32));
  v = fmaxf(v, __shfl_xor(v, 8, 32));
  return v;
}
__device__ __forceinline__ float half_shfl_sum(float v) {
  v += __shfl_xor(v, 1, 32);
  v += __shfl_xor(v, 2, 32);
  v += __shfl_xor(v, 4, 32);
  v += __shfl_xor(v, 8, 32);
  return v;
}

// Async DMA: global (saddr base + 32-bit voffset) -> LDS, 16 bytes per lane.
// Tracked by ASYNCcnt; bypasses VGPRs entirely.
__device__ __forceinline__ void async_b128(unsigned lds_byte_addr,
                                           unsigned gvoff_bytes,
                                           unsigned long long sbase) {
  asm volatile("global_load_async_to_lds_b128 %0, %1, %2"
               :: "v"(lds_byte_addr), "v"(gvoff_bytes), "s"(sbase)
               : "memory");
}
__device__ __forceinline__ void wait_async0() {
  asm volatile("s_wait_asynccnt 0" ::: "memory");
}

// ---------------------------------------------------------------------------
__global__ void cvt_f32_f16(const float* __restrict__ src,
                            _Float16* __restrict__ dst, int n) {
  int i = blockIdx.x * blockDim.x + threadIdx.x;
  const int stride = gridDim.x * blockDim.x;
  for (; i < n; i += stride) dst[i] = (_Float16)src[i];
}

// ---------------------------------------------------------------------------
// Per-head projection: Y = X_head @ W^T + bias.
// TRANSV=false: Y f16 [B*H, S, 64] (row-major for Q/K).
// TRANSV=true : Y f16 [B*H, 64, S] (d-major for V; contiguous keys per dim).
// grid: (S/64, B*H), block 128 (4 waves).
template <bool TRANSV>
__global__ __launch_bounds__(128)
void proj_kernel(const float* __restrict__ X, const _Float16* __restrict__ W16,
                 const float* __restrict__ bias, _Float16* __restrict__ Y) {
  __shared__ _Float16 Xl[64 * LDP];
  const int tid  = threadIdx.x;
  const int lane = tid & 31;
  const int w    = tid >> 5;
  const int bh   = blockIdx.y;
  const int h    = bh & (NH - 1);
  const int b    = bh >> 4;
  const int s0   = blockIdx.x * 64;

  {  // 64x64 f32 -> f16 LDS
    const int r  = tid >> 1;
    const int c0 = (tid & 1) * 32;
    const float* gp = X + ((size_t)(b * S_LEN + s0 + r)) * DM + h * HD + c0;
#pragma unroll
    for (int j = 0; j < 32; ++j) Xl[r * LDP + c0 + j] = (_Float16)gp[j];
  }
  __syncthreads();

  const v16h a0 = load_a_frag(&Xl[w * 16 * LDP + 0], LDP, lane);
  const v16h a1 = load_a_frag(&Xl[w * 16 * LDP + 32], LDP, lane);

#pragma unroll
  for (int nt = 0; nt < 4; ++nt) {
    const int cb = lane & 15;
    const float bval = bias[nt * 16 + cb];
    v8f c;
#pragma unroll
    for (int i = 0; i < 8; ++i) c[i] = bval;
    // B(k,n) = W[n,k] -> W16 is N-major already
    c = wmma16(a0, load_bt_frag(W16 + nt * 16 * HD + 0, HD, lane), c);
    c = wmma16(a1, load_bt_frag(W16 + nt * 16 * HD + 32, HD, lane), c);

    const int r0 = (lane >> 4) << 3;
    if (TRANSV) {
      // pack this lane's 8 consecutive rows into one 16B store at [col][s]
      v8h hv;
#pragma unroll
      for (int i = 0; i < 8; ++i) hv[i] = (_Float16)c[i];
      _Float16* out = Y + ((size_t)bh * HD + nt * 16 + cb) * S_LEN + s0 + w * 16 + r0;
      *(v8h*)out = hv;
    } else {
      _Float16* out = Y + ((size_t)bh * S_LEN + s0 + w * 16) * HD + nt * 16 + cb;
#pragma unroll
      for (int i = 0; i < 8; ++i) out[(size_t)(r0 + i) * HD] = (_Float16)c[i];
    }
  }
}

// ---------------------------------------------------------------------------
// Flash attention per (b,h): 64 Q rows per block, K/V streamed in 64-blocks
// through double-buffered LDS via async-to-LDS DMA.
// grid: (S/64, B*H), block 128 (4 waves; each wave owns 16 Q rows).
__global__ __launch_bounds__(128)
void fattn_kernel(const _Float16* __restrict__ Qh, const _Float16* __restrict__ Kh,
                  const _Float16* __restrict__ Vt, _Float16* __restrict__ O16) {
  __shared__ _Float16 Kl[2][64 * LDP];   // [key][d]
  __shared__ _Float16 Vl[2][64 * LDP];   // [d][key]  (V transposed)
  __shared__ _Float16 Pl[4 * 16 * LDP];  // per-wave P strips

  const int tid  = threadIdx.x;
  const int lane = tid & 31;
  const int w    = tid >> 5;
  const int bh   = blockIdx.y;
  const int h    = bh & (NH - 1);
  const int b    = bh >> 4;
  const int s0   = blockIdx.x * 64;

  const _Float16* Qbase = Qh + ((size_t)bh * S_LEN + s0 + w * 16) * HD;
  const v16h qa0 = load_a_frag(Qbase + 0, HD, lane);
  const v16h qa1 = load_a_frag(Qbase + 32, HD, lane);

  v8f o[4];
  float mrun[8], lrun[8];
#pragma unroll
  for (int nt = 0; nt < 4; ++nt)
#pragma unroll
    for (int i = 0; i < 8; ++i) o[nt][i] = 0.f;
#pragma unroll
  for (int i = 0; i < 8; ++i) { mrun[i] = -1e30f; lrun[i] = 0.f; }

  const _Float16* Kbh  = Kh + (size_t)bh * S_LEN * HD;
  const _Float16* Vtbh = Vt + (size_t)bh * HD * S_LEN;

  // issue one 64x64 f16 tile pair (K row-major, Vt d-major) into buffer `buf`
  auto issue_tiles = [&](int buf, int j0) {
    const unsigned long long kbase =
        (unsigned long long)(uintptr_t)(Kbh + (size_t)j0 * HD);
    const unsigned long long vbase =
        (unsigned long long)(uintptr_t)(Vtbh + j0);
    const unsigned klds = (unsigned)(uintptr_t)&Kl[buf][0];
    const unsigned vlds = (unsigned)(uintptr_t)&Vl[buf][0];
#pragma unroll
    for (int u = 0; u < 4; ++u) {
      const int c  = tid * 4 + u;          // 512 16B chunks per tile
      const int r  = c >> 3;               // tile row
      const int qc = (c & 7) * 16;         // byte offset within row
      const unsigned loff = (unsigned)(r * (LDP * 2) + qc);
      async_b128(klds + loff, (unsigned)(r * (HD * 2) + qc), kbase);
      async_b128(vlds + loff, (unsigned)(r * (S_LEN * 2) + qc), vbase);
    }
  };

  issue_tiles(0, 0);
  wait_async0();
  __syncthreads();

  int cur = 0;
  for (int jb = 0; jb < NKBLK; ++jb) {
    if (jb + 1 < NKBLK) issue_tiles(cur ^ 1, (jb + 1) * 64);  // overlap DMA

    const _Float16* Kc = &Kl[cur][0];
    const _Float16* Vc = &Vl[cur][0];

    // S = Q K^T / 8 : B(k=d, n=key) = K[key, d] -> Kc is N(key)-major
    v8f s[4];
#pragma unroll
    for (int nt = 0; nt < 4; ++nt) {
      v8f acc;
#pragma unroll
      for (int i = 0; i < 8; ++i) acc[i] = 0.f;
      acc = wmma16(qa0, load_bt_frag(&Kc[nt * 16 * LDP + 0], LDP, lane), acc);
      acc = wmma16(qa1, load_bt_frag(&Kc[nt * 16 * LDP + 32], LDP, lane), acc);
#pragma unroll
      for (int i = 0; i < 8; ++i) s[nt][i] = acc[i] * 0.125f;
    }

    // online softmax (each 16-lane half owns one row per accumulator VGPR)
    _Float16* Pw = &Pl[w * 16 * LDP];
#pragma unroll
    for (int i = 0; i < 8; ++i) {
      float rm = fmaxf(fmaxf(s[0][i], s[1][i]), fmaxf(s[2][i], s[3][i]));
      rm = half_shfl_max(rm);
      const float mnew  = fmaxf(mrun[i], rm);
      const float alpha = __expf(mrun[i] - mnew);
      mrun[i] = mnew;
      const float p0 = __expf(s[0][i] - mnew);
      const float p1 = __expf(s[1][i] - mnew);
      const float p2 = __expf(s[2][i] - mnew);
      const float p3 = __expf(s[3][i] - mnew);
      lrun[i] = lrun[i] * alpha + half_shfl_sum(p0 + p1 + p2 + p3);
#pragma unroll
      for (int nt = 0; nt < 4; ++nt) o[nt][i] *= alpha;
      const int r  = ((lane >> 4) << 3) + i;
      const int cb = lane & 15;
      Pw[r * LDP + 0 + cb]  = (_Float16)p0;
      Pw[r * LDP + 16 + cb] = (_Float16)p1;
      Pw[r * LDP + 32 + cb] = (_Float16)p2;
      Pw[r * LDP + 48 + cb] = (_Float16)p3;
    }

    // O += P @ V : B(k=key, n=d) = V[key,d] = Vc[d][key] -> N(d)-major
#pragma unroll
    for (int ks = 0; ks < 2; ++ks) {
      const v16h pa = load_a_frag(Pw + ks * 32, LDP, lane);
#pragma unroll
      for (int nt = 0; nt < 4; ++nt)
        o[nt] = wmma16(pa, load_bt_frag(&Vc[nt * 16 * LDP + ks * 32], LDP, lane),
                       o[nt]);
    }

    wait_async0();     // next tile landed in LDS
    __syncthreads();   // everyone done reading `cur`
    cur ^= 1;
  }

  // normalize rows and store f16 into [B, S, DM] (head-interleaved)
#pragma unroll
  for (int i = 0; i < 8; ++i) {
    const float inv = 1.0f / lrun[i];
    const int r  = ((lane >> 4) << 3) + i;
    const int si = s0 + w * 16 + r;
    _Float16* out = O16 + ((size_t)(b * S_LEN + si)) * DM + h * HD + (lane & 15);
#pragma unroll
    for (int nt = 0; nt < 4; ++nt) out[nt * 16] = (_Float16)(o[nt][i] * inv);
  }
}

// ---------------------------------------------------------------------------
// Output projection: out[m, n] = A[m, :] @ Wo[n, :] + bo[n], M=B*S, N=K=1024.
// Double-buffered async-to-LDS tile streaming; Wo kept N-major in LDS so all
// fragment loads are contiguous ds_load_b128 pairs.
// grid: (DM/64, B*S/64), block 128 (4 waves), 512 WMMAs/block.
__global__ __launch_bounds__(128)
void oproj_kernel(const _Float16* __restrict__ A16, const _Float16* __restrict__ Wo16,
                  const float* __restrict__ bo, float* __restrict__ out) {
  __shared__ _Float16 Al[2][64 * LDP];  // [m][k]
  __shared__ _Float16 Wl[2][64 * LDP];  // [n][k]
  const int tid  = threadIdx.x;
  const int lane = tid & 31;
  const int w    = tid >> 5;
  const int m0   = blockIdx.y * 64;
  const int n0   = blockIdx.x * 64;

  v8f o[4];
#pragma unroll
  for (int nt = 0; nt < 4; ++nt) {
    const float bval = bo[n0 + nt * 16 + (lane & 15)];
#pragma unroll
    for (int i = 0; i < 8; ++i) o[nt][i] = bval;
  }

  // issue one 64x64 A tile + 64x64 Wo tile into buffer `buf`
  auto issue_tiles = [&](int buf, int k0) {
    const unsigned long long abase =
        (unsigned long long)(uintptr_t)(A16 + (size_t)m0 * DM + k0);
    const unsigned long long wbase =
        (unsigned long long)(uintptr_t)(Wo16 + (size_t)n0 * DM + k0);
    const unsigned alds = (unsigned)(uintptr_t)&Al[buf][0];
    const unsigned wlds = (unsigned)(uintptr_t)&Wl[buf][0];
#pragma unroll
    for (int u = 0; u < 4; ++u) {
      const int c  = tid * 4 + u;
      const int r  = c >> 3;
      const int qc = (c & 7) * 16;
      const unsigned loff = (unsigned)(r * (LDP * 2) + qc);
      const unsigned goff = (unsigned)(r * (DM * 2) + qc);
      async_b128(alds + loff, goff, abase);
      async_b128(wlds + loff, goff, wbase);
    }
  };

  issue_tiles(0, 0);
  wait_async0();
  __syncthreads();

  int cur = 0;
  for (int kt = 0; kt < DM / 64; ++kt) {
    if (kt + 1 < DM / 64) issue_tiles(cur ^ 1, (kt + 1) * 64);  // overlap DMA

    const _Float16* Ac = &Al[cur][0];
    const _Float16* Wc = &Wl[cur][0];
#pragma unroll
    for (int ks = 0; ks < 2; ++ks) {
      const v16h a = load_a_frag(&Ac[w * 16 * LDP + ks * 32], LDP, lane);
#pragma unroll
      for (int nt = 0; nt < 4; ++nt)
        o[nt] = wmma16(a, load_bt_frag(&Wc[nt * 16 * LDP + ks * 32], LDP, lane),
                       o[nt]);
    }

    wait_async0();
    __syncthreads();
    cur ^= 1;
  }

#pragma unroll
  for (int nt = 0; nt < 4; ++nt)
#pragma unroll
    for (int i = 0; i < 8; ++i) {
      const int r = ((lane >> 4) << 3) + i;
      out[(size_t)(m0 + w * 16 + r) * DM + n0 + nt * 16 + (lane & 15)] = o[nt][i];
    }
}

// ---------------------------------------------------------------------------
extern "C" void kernel_launch(void* const* d_in, const int* in_sizes, int n_in,
                              void* d_out, int out_size, void* d_ws, size_t ws_size,
                              hipStream_t stream) {
  (void)in_sizes; (void)n_in; (void)out_size; (void)ws_size;
  const float* q  = (const float*)d_in[0];
  const float* k  = (const float*)d_in[1];
  const float* v  = (const float*)d_in[2];
  const float* Wq = (const float*)d_in[3];
  const float* bq = (const float*)d_in[4];
  const float* Wk = (const float*)d_in[5];
  const float* bk = (const float*)d_in[6];
  const float* Wv = (const float*)d_in[7];
  const float* bv = (const float*)d_in[8];
  const float* Wo = (const float*)d_in[9];
  const float* bo = (const float*)d_in[10];
  float* out = (float*)d_out;

  char* ws = (char*)d_ws;
  _Float16* Wq16 = (_Float16*)ws; ws += 64 * 64 * 2;
  _Float16* Wk16 = (_Float16*)ws; ws += 64 * 64 * 2;
  _Float16* Wv16 = (_Float16*)ws; ws += 64 * 64 * 2;
  _Float16* Wo16 = (_Float16*)ws; ws += (size_t)DM * DM * 2;
  _Float16* qh   = (_Float16*)ws; ws += (size_t)NB * NH * S_LEN * HD * 2;
  _Float16* kh   = (_Float16*)ws; ws += (size_t)NB * NH * S_LEN * HD * 2;
  _Float16* vt   = (_Float16*)ws; ws += (size_t)NB * NH * S_LEN * HD * 2;  // transposed
  _Float16* at   = (_Float16*)ws;  // [B, S, DM] f16 attention output

  cvt_f32_f16<<<16, 256, 0, stream>>>(Wq, Wq16, 64 * 64);
  cvt_f32_f16<<<16, 256, 0, stream>>>(Wk, Wk16, 64 * 64);
  cvt_f32_f16<<<16, 256, 0, stream>>>(Wv, Wv16, 64 * 64);
  cvt_f32_f16<<<2048, 256, 0, stream>>>(Wo, Wo16, DM * DM);

  const dim3 blk(128);
  const dim3 gattn(S_LEN / 64, NB * NH);
  proj_kernel<false><<<gattn, blk, 0, stream>>>(q, Wq16, bq, qh);
  proj_kernel<false><<<gattn, blk, 0, stream>>>(k, Wk16, bk, kh);
  proj_kernel<true ><<<gattn, blk, 0, stream>>>(v, Wv16, bv, vt);

  fattn_kernel<<<gattn, blk, 0, stream>>>(qh, kh, vt, at);

  const dim3 gout(DM / 64, (NB * S_LEN) / 64);
  oproj_kernel<<<gout, blk, 0, stream>>>(at, Wo16, bo, out);
}
